// SkeletonGraphAttention_66099546685478
// MI455X (gfx1250) — matrix operand order repaired and checked by reading, not compiled, not run on hardware
//
#include <hip/hip_runtime.h>
#include <hip/hip_bf16.h>
#include <math.h>

typedef __attribute__((ext_vector_type(16))) _Float16 v16h;
typedef __attribute__((ext_vector_type(8)))  float    v8f;

#define Bgl   8192
#define Nn    25
#define DIN   256
#define Hh    8
#define Ff    32
#define Eb    48
#define Et    (Eb + Nn)       // 73 edges incl self loops
#define BB    16              // batches per workgroup
#define NEG_SLOPE 0.2f

__global__ __launch_bounds__(256) void
skeleton_gat_fused(const float* __restrict__ Xg,      // (B, N, 256)
                   const float* __restrict__ Wg,      // (256, 256)
                   const float* __restrict__ aSrcG,   // (8, 32)
                   const float* __restrict__ aDstG,   // (8, 32)
                   const float* __restrict__ biasG,   // (256,)
                   const int*   __restrict__ edgesG,  // (2, 48)
                   float*       __restrict__ outG)    // (B, N, 256)
{
    // ---------------- LDS ----------------
    __shared__ _Float16 Hlds[Nn * BB * DIN];          // 204800 B : h tile (f16)
    __shared__ _Float16 Xlds[BB * DIN];               //   8192 B : X tile for one node
    __shared__ float    alphaS[BB * Nn * Hh];         //  12800 B
    __shared__ float    alphaD[BB * Nn * Hh];         //  12800 B
    __shared__ float    eL[BB * Et * Hh];             //  37376 B : leaky-relu'd logits
    __shared__ float    aS[Hh * Ff], aD[Hh * Ff], biasL[DIN];
    __shared__ int      srcE[Et], dstE[Et];

    const int t    = threadIdx.x;
    const int lane = t & 31;
    const int wave = t >> 5;                          // 8 waves = 8 heads
    const int b0   = blockIdx.x * BB;

    // ---------------- phase 0: small constants ----------------
    aS[t]    = aSrcG[t];
    aD[t]    = aDstG[t];
    biasL[t] = biasG[t];
    if (t < Eb)            { srcE[t] = edgesG[t]; dstE[t] = edgesG[Eb + t]; }
    else if (t < Et)       { srcE[t] = t - Eb;    dstE[t] = t - Eb;         }

    // ---------------- phase 0b: persistent B fragments (W cols, f32->f16) ----
    // B (K x N = 32 x 16, 16-bit): lanes 0-15 col N=lane, K=0..15;
    //                              lanes 16-31 col N=lane-16, K=16..31.
    const int colInTile = lane & 15;
    const int kHalfB    = (lane < 16) ? 0 : 16;
    v16h bfrag[2][8];
#pragma unroll
    for (int nt = 0; nt < 2; ++nt) {
        const int c = wave * 32 + nt * 16 + colInTile;
#pragma unroll
        for (int kk = 0; kk < 8; ++kk) {
#pragma unroll
            for (int i = 0; i < 16; ++i) {
                const int k = kk * 32 + kHalfB + i;
                bfrag[nt][kk][i] = (_Float16)Wg[k * DIN + c];
            }
        }
    }

    // ---------------- phase 1: GEMM  h = X @ W  (per node, M = batch) -------
    // A (16-bit 16x32): lanes 0-15 M=lane, j=0..15 -> K=j
    //                   lanes 16-31: K=16+j.
    const int mrow  = lane & 15;
    const int aOff  = (lane < 16) ? 0 : 16;
    const int rbase = (lane < 16) ? 0 : 8;            // C rows: M = r + rbase

    for (int n = 0; n < Nn; ++n) {
        __syncthreads();                               // Xlds reuse guard
        for (int idx = t; idx < BB * DIN; idx += 256) {
            const int m = idx >> 8, k = idx & 255;
            Xlds[idx] = (_Float16)Xg[((size_t)(b0 + m) * Nn + n) * DIN + k];
        }
        __syncthreads();

        v8f c0 = {}, c1 = {};
#pragma unroll
        for (int kk = 0; kk < 8; ++kk) {
            v16h afrag;
#pragma unroll
            for (int j = 0; j < 8; ++j)
                afrag[j] = Xlds[mrow * DIN + kk * 32 + aOff + j];
#pragma unroll
            for (int j = 0; j < 8; ++j)
                afrag[8 + j] = Xlds[mrow * DIN + kk * 32 + 8 + aOff + j];
            c0 = __builtin_amdgcn_wmma_f32_16x16x32_f16(false, afrag, false, bfrag[0][kk],
                                                        (short)0, c0, false, false);
            c1 = __builtin_amdgcn_wmma_f32_16x16x32_f16(false, afrag, false, bfrag[1][kk],
                                                        (short)0, c1, false, false);
        }
        // C layout: VGPR r -> M = rbase + r, N = lane&15
#pragma unroll
        for (int r = 0; r < 8; ++r) {
            const int mm = rbase + r;
            Hlds[(n * BB + mm) * DIN + wave * 32 + colInTile]      = (_Float16)c0[r];
            Hlds[(n * BB + mm) * DIN + wave * 32 + 16 + colInTile] = (_Float16)c1[r];
        }
    }
    __syncthreads();

    // ---------------- phase 2: per-node attention logits --------------------
    for (int idx = t; idx < BB * Nn * Hh; idx += 256) {
        const int h_ = idx & 7;
        const int n_ = (idx >> 3) % Nn;
        const int b_ = idx / (Nn * Hh);
        const _Float16* hp = &Hlds[(n_ * BB + b_) * DIN + h_ * Ff];
        float s = 0.f, d = 0.f;
#pragma unroll
        for (int j = 0; j < Ff; ++j) {
            const float v = (float)hp[j];
            s += v * aS[h_ * Ff + j];
            d += v * aD[h_ * Ff + j];
        }
        alphaS[idx] = s;
        alphaD[idx] = d;
    }
    __syncthreads();

    // ---------------- phase 2b: per-edge leaky-relu logits -------------------
    for (int idx = t; idx < BB * Et * Hh; idx += 256) {
        const int h_ = idx & 7;
        const int e_ = (idx >> 3) % Et;
        const int b_ = idx / (Et * Hh);
        const float x = alphaS[(b_ * Nn + srcE[e_]) * Hh + h_]
                      + alphaD[(b_ * Nn + dstE[e_]) * Hh + h_];
        eL[idx] = (x > 0.f) ? x : NEG_SLOPE * x;       // idx == (b*Et+e)*Hh+h
    }
    __syncthreads();

    // ---------------- phase 3: segment softmax + aggregate -------------------
    // thread -> (batch, head, half-of-F): 16 b x 8 h x 2 halves = 256
    {
        const int b_  = t >> 4;
        const int h_  = (t >> 1) & 7;
        const int f0  = (t & 1) * 16;

        for (int d = 0; d < Nn; ++d) {
            float m = -1e30f;                          // self-loop guarantees an edge
            for (int e = 0; e < Et; ++e)
                if (dstE[e] == d) m = fmaxf(m, eL[(b_ * Et + e) * Hh + h_]);
            float denom = 0.f;
            for (int e = 0; e < Et; ++e)
                if (dstE[e] == d) denom += __expf(eL[(b_ * Et + e) * Hh + h_] - m);
            const float inv = 1.0f / (denom + 1e-16f);

            float acc[16];
#pragma unroll
            for (int j = 0; j < 16; ++j) acc[j] = 0.f;
            for (int e = 0; e < Et; ++e) {
                if (dstE[e] != d) continue;
                const float w = __expf(eL[(b_ * Et + e) * Hh + h_] - m) * inv;
                const _Float16* hp = &Hlds[(srcE[e] * BB + b_) * DIN + h_ * Ff + f0];
#pragma unroll
                for (int j = 0; j < 16; ++j) acc[j] += w * (float)hp[j];
            }
            float* op = outG + (((size_t)(b0 + b_) * Nn + d) * DIN + h_ * Ff + f0);
#pragma unroll
            for (int j = 0; j < 16; ++j) op[j] = acc[j] + biasL[h_ * Ff + f0 + j];
        }
    }
}

extern "C" void kernel_launch(void* const* d_in, const int* in_sizes, int n_in,
                              void* d_out, int out_size, void* d_ws, size_t ws_size,
                              hipStream_t stream) {
    const float* X     = (const float*)d_in[0];
    const float* W     = (const float*)d_in[1];
    const float* aSrc  = (const float*)d_in[2];
    const float* aDst  = (const float*)d_in[3];
    const float* bias  = (const float*)d_in[4];
    const int*   edges = (const int*)d_in[5];
    float*       out   = (float*)d_out;
    (void)in_sizes; (void)n_in; (void)out_size; (void)d_ws; (void)ws_size;

    skeleton_gat_fused<<<Bgl / BB, 256, 0, stream>>>(X, W, aSrc, aDst, bias, edges, out);
}
